// SimpleGateLoopLayer_82875688944201
// MI455X (gfx1250) — matrix operand order repaired and checked by reading, not compiled, not run on hardware
//
#include <hip/hip_runtime.h>
#include <hip/hip_bf16.h>
#include <math.h>

typedef __bf16 bf16;
typedef __attribute__((ext_vector_type(16))) __bf16 v16bf;
typedef __attribute__((ext_vector_type(8)))  __bf16 v8bf;
typedef __attribute__((ext_vector_type(8)))  float  v8f;
typedef int v4i_gcc __attribute__((vector_size(16)));

#define D_DIM 1024
#define N_SEQ 4096
#define B_SZ  4
#define M_TOT (B_SZ * N_SEQ)   // 16384 rows
#define N_OUT (3 * D_DIM)      // 3072 cols
#define NCHUNK 16
#define CLEN  (N_SEQ / NCHUNK) // 256

// ---- CDNA5 async global->LDS copy (ASYNCcnt path), with safe fallback ----
__device__ __forceinline__ void async_copy16(const bf16* g, bf16* l) {
#if __has_builtin(__builtin_amdgcn_global_load_async_to_lds_b128)
    __builtin_amdgcn_global_load_async_to_lds_b128(
        (__attribute__((address_space(1))) v4i_gcc*)(void*)g,
        (__attribute__((address_space(3))) v4i_gcc*)(void*)l, 0, 0);
#else
    *(uint4*)l = *(const uint4*)g;
#endif
}

__device__ __forceinline__ void wait_async_all() {
#if __has_builtin(__builtin_amdgcn_s_wait_asynccnt)
    __builtin_amdgcn_s_wait_asynccnt(0);
#endif
}

// ------------------------------------------------------------------
// K0: W [D, 3D] f32 (row-major, k-major) -> Wt [3D, D] bf16
// ------------------------------------------------------------------
__global__ void wcast_kernel(const float* __restrict__ w, bf16* __restrict__ wt) {
    int idx = blockIdx.x * 256 + threadIdx.x;
    if (idx >= D_DIM * N_OUT) return;
    int k = idx / N_OUT;
    int n = idx % N_OUT;
    wt[(size_t)n * D_DIM + k] = (bf16)w[idx];
}

// ------------------------------------------------------------------
// K1: RMSNorm + cast to bf16. One 256-thread block per row.
// ------------------------------------------------------------------
__global__ void rmsnorm_kernel(const float* __restrict__ x,
                               const float* __restrict__ gamma,
                               bf16* __restrict__ xn) {
    __shared__ float red[256];
    const int row = blockIdx.x;
    const int tid = threadIdx.x;
    const float* xr = x + (size_t)row * D_DIM;

    float v[4];
    float s = 0.f;
#pragma unroll
    for (int i = 0; i < 4; i++) {
        v[i] = xr[tid + i * 256];
        s += v[i] * v[i];
    }
    red[tid] = s;
    __syncthreads();
    for (int off = 128; off > 0; off >>= 1) {
        if (tid < off) red[tid] += red[tid + off];
        __syncthreads();
    }
    const float norm  = sqrtf(red[0]);
    const float scale = 32.0f / fmaxf(norm, 1e-12f);  // sqrt(1024) = 32

    bf16* xo = xn + (size_t)row * D_DIM;
#pragma unroll
    for (int i = 0; i < 4; i++) {
        int c = tid + i * 256;
        xo[c] = (bf16)(v[i] * scale * gamma[c]);
    }
}

// ------------------------------------------------------------------
// K2: bf16 WMMA GEMM  [M_TOT x 1024] x [1024 x 3072]
// Block tile 128(M) x 128(N), 8 waves (4x2), wave tile 32x64.
// Double-buffered LDS, async global->LDS staging, 8 WMMA / K-step.
// ------------------------------------------------------------------
__global__ __launch_bounds__(256) void gemm_qkva_kernel(
        const bf16* __restrict__ xn, const bf16* __restrict__ wt,
        float* __restrict__ Q, float* __restrict__ KV, float* __restrict__ Ag) {
    // padded strides: A row 80B (16B-aligned frags), B row 96B (32B-aligned frags)
    __shared__ __attribute__((aligned(64))) bf16 lds_a[2][128][40];
    __shared__ __attribute__((aligned(64))) bf16 lds_b[2][128][48];

    const int tid   = threadIdx.x;
    const int lane  = tid & 31;
    const int wid   = tid >> 5;
    const int waveM = wid & 3;        // 0..3
    const int waveN = wid >> 2;       // 0..1
    const int mBase = blockIdx.y * 128;
    const int nBase = blockIdx.x * 128;
    const int wm    = waveM * 32;
    const int wn    = waveN * 64;
    const int hf    = lane >> 4;      // lane half (0/1)
    const int l16   = lane & 15;

    // staging indices: each thread moves 2 A-chunks + 2 B-chunks of 16B
    const int sr0 = tid >> 2,           sq0 = tid & 3;          // chunks 0..255
    const int sr1 = (tid + 256) >> 2,   sq1 = (tid + 256) & 3;  // chunks 256..511

    auto stage = [&](int k0, int bi) {
        async_copy16(xn + (size_t)(mBase + sr0) * D_DIM + k0 + sq0 * 8, &lds_a[bi][sr0][sq0 * 8]);
        async_copy16(xn + (size_t)(mBase + sr1) * D_DIM + k0 + sq1 * 8, &lds_a[bi][sr1][sq1 * 8]);
        async_copy16(wt + (size_t)(nBase + sr0) * D_DIM + k0 + sq0 * 8, &lds_b[bi][sr0][sq0 * 8]);
        async_copy16(wt + (size_t)(nBase + sr1) * D_DIM + k0 + sq1 * 8, &lds_b[bi][sr1][sq1 * 8]);
    };

    v8f acc[2][4] = {};

    int buf = 0;
    stage(0, 0);
    wait_async_all();
    __syncthreads();

    for (int k0 = 0; k0 < D_DIM; k0 += 32) {
        const bool more = (k0 + 32) < D_DIM;
        if (more) stage(k0 + 32, buf ^ 1);   // async into other buffer

        // Fragments, contiguous k = 16*hf + s in both A and B (16-bit A 16x32, B 32x16)
        v16bf afrag[2], bfrag[4];
#pragma unroll
        for (int t = 0; t < 2; t++) {
            int m = wm + t * 16 + l16;
            v8bf lo = *(const v8bf*)&lds_a[buf][m][hf * 16];
            v8bf hi = *(const v8bf*)&lds_a[buf][m][hf * 16 + 8];
            v16bf f;
#pragma unroll
            for (int i = 0; i < 8; i++) { f[i] = lo[i]; f[8 + i] = hi[i]; }
            afrag[t] = f;
        }
#pragma unroll
        for (int t = 0; t < 4; t++) {
            int n = wn + t * 16 + l16;
            bfrag[t] = *(const v16bf*)&lds_b[buf][n][hf * 16];
        }
#pragma unroll
        for (int tm = 0; tm < 2; tm++)
#pragma unroll
            for (int tn = 0; tn < 4; tn++)
                acc[tm][tn] = __builtin_amdgcn_wmma_f32_16x16x32_bf16(
                    false, afrag[tm], false, bfrag[tn],
                    (short)0, acc[tm][tn], false, false);

        if (more) {
            wait_async_all();     // own async stores to LDS done
            __syncthreads();      // everyone's done; safe to read next buf
            buf ^= 1;
        }
    }

    // Epilogue: D layout = VGPR j -> M = j + 8*half, N = l16
    const int region = nBase >> 10;  // uniform per block (128 | 1024)
#pragma unroll
    for (int tm = 0; tm < 2; tm++) {
#pragma unroll
        for (int tn = 0; tn < 4; tn++) {
            int colG = nBase + wn + tn * 16 + l16;
            int cin  = colG & 1023;
#pragma unroll
            for (int j = 0; j < 8; j++) {
                int mG = mBase + wm + tm * 16 + j + hf * 8;
                size_t off = (size_t)mG * D_DIM + cin;
                float v = acc[tm][tn][j];
                if (region == 0)      Q[off]  = v;
                else if (region == 1) KV[off] = v;
                else                  Ag[off] = 1.0f / (1.0f + __expf(-v));
            }
        }
    }
}

// ------------------------------------------------------------------
// K3: per-chunk summaries: Aprod = prod(a), Kvsum = scan tail (zero init)
// ------------------------------------------------------------------
__global__ void chunk_summary_kernel(const float* __restrict__ Ag,
                                     const float* __restrict__ KV,
                                     float* __restrict__ Aprod,
                                     float* __restrict__ Kvsum) {
    int idx = blockIdx.x * 256 + threadIdx.x;     // 65536
    int d = idx & (D_DIM - 1);
    int b = (idx >> 10) & (B_SZ - 1);
    int c = idx >> 12;                            // 0..15
    size_t base = ((size_t)b * N_SEQ + (size_t)c * CLEN) * D_DIM + d;
    float ap = 1.f, kv = 0.f;
    for (int t = 0; t < CLEN; t++) {
        size_t o = base + (size_t)t * D_DIM;
        float a = Ag[o];
        kv = a * kv + KV[o];
        ap *= a;
    }
    int so = (b * NCHUNK + c) * D_DIM + d;
    Aprod[so] = ap;
    Kvsum[so] = kv;
}

// ------------------------------------------------------------------
// K4: sequential scan over the 16 chunk summaries per (b,d) channel
// ------------------------------------------------------------------
__global__ void chunk_carry_kernel(const float* __restrict__ Aprod,
                                   const float* __restrict__ Kvsum,
                                   float* __restrict__ Carry) {
    int idx = blockIdx.x * 256 + threadIdx.x;     // 4096
    int d = idx & (D_DIM - 1);
    int b = idx >> 10;
    float carry = 0.f;
    for (int c = 0; c < NCHUNK; c++) {
        int so = (b * NCHUNK + c) * D_DIM + d;
        Carry[so] = carry;
        carry = Aprod[so] * carry + Kvsum[so];
    }
}

// ------------------------------------------------------------------
// K5: apply carries, redo local scan, out = q * kv
// ------------------------------------------------------------------
__global__ void chunk_apply_kernel(const float* __restrict__ Ag,
                                   const float* __restrict__ KV,
                                   const float* __restrict__ Q,
                                   const float* __restrict__ Carry,
                                   float* __restrict__ out) {
    int idx = blockIdx.x * 256 + threadIdx.x;     // 65536
    int d = idx & (D_DIM - 1);
    int b = (idx >> 10) & (B_SZ - 1);
    int c = idx >> 12;
    size_t base = ((size_t)b * N_SEQ + (size_t)c * CLEN) * D_DIM + d;
    float kv = Carry[(b * NCHUNK + c) * D_DIM + d];
    for (int t = 0; t < CLEN; t++) {
        size_t o = base + (size_t)t * D_DIM;
        kv = Ag[o] * kv + KV[o];
        out[o] = Q[o] * kv;
    }
}

// ------------------------------------------------------------------
extern "C" void kernel_launch(void* const* d_in, const int* in_sizes, int n_in,
                              void* d_out, int out_size, void* d_ws, size_t ws_size,
                              hipStream_t stream) {
    const float* x     = (const float*)d_in[0];   // [B, N, D]
    const float* w     = (const float*)d_in[1];   // [D, 3D]
    const float* gamma = (const float*)d_in[2];   // [D]
    float* out = (float*)d_out;                   // [B, N, D] f32

    char* ws = (char*)d_ws;
    size_t off = 0;
    auto carve = [&](size_t bytes) -> void* {
        void* p = ws + off;
        off = (off + bytes + 255) & ~(size_t)255;
        return p;
    };

    bf16*  Wt    = (bf16*) carve((size_t)N_OUT * D_DIM * sizeof(bf16));   // 6 MB
    bf16*  Xn    = (bf16*) carve((size_t)M_TOT * D_DIM * sizeof(bf16));   // 32 MB
    float* Q     = (float*)carve((size_t)M_TOT * D_DIM * sizeof(float));  // 64 MB
    float* KV    = (float*)carve((size_t)M_TOT * D_DIM * sizeof(float));  // 64 MB
    float* Ag    = (float*)carve((size_t)M_TOT * D_DIM * sizeof(float));  // 64 MB
    float* Aprod = (float*)carve((size_t)B_SZ * NCHUNK * D_DIM * sizeof(float));
    float* Kvsum = (float*)carve((size_t)B_SZ * NCHUNK * D_DIM * sizeof(float));
    float* Carry = (float*)carve((size_t)B_SZ * NCHUNK * D_DIM * sizeof(float));
    (void)ws_size; (void)in_sizes; (void)n_in; (void)out_size;

    wcast_kernel<<<(D_DIM * N_OUT + 255) / 256, 256, 0, stream>>>(w, Wt);
    rmsnorm_kernel<<<M_TOT, 256, 0, stream>>>(x, gamma, Xn);
    dim3 gemm_grid(N_OUT / 128, M_TOT / 128);   // 24 x 128
    gemm_qkva_kernel<<<gemm_grid, 256, 0, stream>>>(Xn, Wt, Q, KV, Ag);
    chunk_summary_kernel<<<(B_SZ * D_DIM * NCHUNK) / 256, 256, 0, stream>>>(Ag, KV, Aprod, Kvsum);
    chunk_carry_kernel<<<(B_SZ * D_DIM) / 256, 256, 0, stream>>>(Aprod, Kvsum, Carry);
    chunk_apply_kernel<<<(B_SZ * D_DIM * NCHUNK) / 256, 256, 0, stream>>>(Ag, KV, Q, Carry, out);
}
